// Block_65996467470380
// MI455X (gfx1250) — compile-verified, matrix operand order checked by reading, not run on hardware
//
#include <hip/hip_runtime.h>
#include <hip/hip_bf16.h>
#include <math.h>

typedef __attribute__((ext_vector_type(16))) __bf16 v16bf;
typedef __attribute__((ext_vector_type(8)))  float  v8f;

#define BB 2
#define TT 2048
#define CC 1024
#define HH 16
#define HD 64

static __device__ inline v8f wmma_bf16(v16bf a, v16bf b, v8f c) {
  return __builtin_amdgcn_wmma_f32_16x16x32_bf16(false, a, false, b, (short)0, c, false, false);
}

static __device__ inline float gelu_tanh(float x) {
  const float c = 0.7978845608028654f;
  return 0.5f * x * (1.0f + tanhf(c * (x + 0.044715f * x * x * x)));
}

// A/B 16x32 bf16 fragment from a row-major bf16 tile (LDS or global).
static __device__ inline v16bf frag_rowmajor(const __bf16* p, int ld, int lane) {
  const int m  = lane & 15;
  const int ko = (lane >> 4) << 3;
  const __bf16* q = p + (size_t)m * ld + ko;
  v16bf f;
#pragma unroll
  for (int j = 0; j < 4; ++j) {
    f[2 * j]     = q[2 * j];
    f[2 * j + 1] = q[2 * j + 1];
    f[8 + 2 * j] = q[16 + 2 * j];
    f[9 + 2 * j] = q[16 + 2 * j + 1];
  }
  return f;
}

// B fragment where the K dimension strides by ldk (V tile: keys x d).
static __device__ inline v16bf frag_b_strided(const __bf16* p, int ldk, int lane) {
  const int n  = lane & 15;
  const int ko = (lane >> 4) << 3;
  const __bf16* q = p + n;
  v16bf f;
#pragma unroll
  for (int j = 0; j < 4; ++j) {
    f[2 * j]     = q[(size_t)(ko + 2 * j) * ldk];
    f[2 * j + 1] = q[(size_t)(ko + 2 * j + 1) * ldk];
    f[8 + 2 * j] = q[(size_t)(16 + ko + 2 * j) * ldk];
    f[9 + 2 * j] = q[(size_t)(16 + ko + 2 * j + 1) * ldk];
  }
  return f;
}

// ---- f32 -> bf16 conversion (weights, once per launch) ---------------------
__global__ void wconv_kernel(const float* __restrict__ src, __bf16* __restrict__ dst, int n) {
  const int i = blockIdx.x * blockDim.x + threadIdx.x;
  if (i < n) dst[i] = (__bf16)src[i];
}

// ---- LayerNorm: one 256-thread block per row of C=1024, bf16 out -----------
__global__ void ln_kernel(const float* __restrict__ x, const float* __restrict__ w,
                          const float* __restrict__ b, __bf16* __restrict__ out) {
  const int row = blockIdx.x;
  const int tid = threadIdx.x;
  const float* xr = x + (size_t)row * CC;
  float v[4];
  float s = 0.f, sq = 0.f;
#pragma unroll
  for (int i = 0; i < 4; ++i) {
    v[i] = xr[tid + i * 256];
    s += v[i];
    sq += v[i] * v[i];
  }
#pragma unroll
  for (int off = 16; off >= 1; off >>= 1) {
    s  += __shfl_xor(s,  off, 32);
    sq += __shfl_xor(sq, off, 32);
  }
  __shared__ float reds[8], redq[8];
  const int wid = tid >> 5;
  if ((tid & 31) == 0) { reds[wid] = s; redq[wid] = sq; }
  __syncthreads();
  float ts = 0.f, tq = 0.f;
#pragma unroll
  for (int k = 0; k < 8; ++k) { ts += reds[k]; tq += redq[k]; }
  const float mean = ts * (1.0f / CC);
  const float var  = tq * (1.0f / CC) - mean * mean;
  const float rstd = rsqrtf(var + 1e-5f);
  __bf16* orow = out + (size_t)row * CC;
#pragma unroll
  for (int i = 0; i < 4; ++i) {
    const int c = tid + i * 256;
    orow[c] = (__bf16)((v[i] - mean) * rstd * w[c] + b[c]);
  }
}

// ---- WMMA GEMM: out[M,N] = A[M,K](bf16) @ W[N,K]^T(bf16) + bias ------------
// Epilogue modes: 0 = f32 out + residual add, 1 = bf16 out, 2 = gelu -> bf16.
// 256 threads = 8 waves, block tile 128x128, wave tile 64x32, K-step 32.
template <int EM>
__global__ void __launch_bounds__(256, 1)
gemm_kernel(const __bf16* __restrict__ A, const __bf16* __restrict__ W,
            const float* __restrict__ bias, const float* __restrict__ resid,
            void* __restrict__ outp, int M, int N, int K) {
  __shared__ __bf16 As[128 * 40];
  __shared__ __bf16 Bs[128 * 40];
  const int tid  = threadIdx.x;
  const int lane = tid & 31;
  const int wid  = tid >> 5;
  const int wm   = wid & 1;   // 0..1 -> 64-row slab
  const int wn   = wid >> 1;  // 0..3 -> 32-col slab
  const int m0   = blockIdx.y * 128;
  const int n0   = blockIdx.x * 128;

  v8f acc[4][2] = {};

  const int srow = tid >> 1, sseg = tid & 1;  // 16 bf16 (32B) per thread per tile
  const __bf16* agp = A + (size_t)(m0 + srow) * K + sseg * 16;
  const __bf16* bgp = W + (size_t)(n0 + srow) * K + sseg * 16;

  for (int k0 = 0; k0 < K; k0 += 32) {
    {
      const uint4* ap = (const uint4*)(agp + k0);
      uint4* ad = (uint4*)(As + srow * 40 + sseg * 16);
      ad[0] = ap[0]; ad[1] = ap[1];
      const uint4* bp = (const uint4*)(bgp + k0);
      uint4* bd = (uint4*)(Bs + srow * 40 + sseg * 16);
      bd[0] = bp[0]; bd[1] = bp[1];
    }
    if (k0 + 32 < K) {  // prefetch next K-step tiles into cache (global_prefetch_b8)
      __builtin_prefetch(agp + k0 + 32, 0, 0);
      __builtin_prefetch(bgp + k0 + 32, 0, 0);
    }
    __syncthreads();

    v16bf af[4], bf[2];
#pragma unroll
    for (int i = 0; i < 4; ++i) af[i] = frag_rowmajor(As + (wm * 64 + i * 16) * 40, 40, lane);
#pragma unroll
    for (int j = 0; j < 2; ++j) bf[j] = frag_rowmajor(Bs + (wn * 32 + j * 16) * 40, 40, lane);
#pragma unroll
    for (int i = 0; i < 4; ++i)
#pragma unroll
      for (int j = 0; j < 2; ++j) acc[i][j] = wmma_bf16(af[i], bf[j], acc[i][j]);
    __syncthreads();
  }

  const int hi = lane >> 4;
  float biasj[2];
#pragma unroll
  for (int j = 0; j < 2; ++j) biasj[j] = bias[n0 + wn * 32 + j * 16 + (lane & 15)];

#pragma unroll
  for (int i = 0; i < 4; ++i)
#pragma unroll
    for (int j = 0; j < 2; ++j)
#pragma unroll
      for (int r = 0; r < 8; ++r) {
        const int row = m0 + wm * 64 + i * 16 + r + 8 * hi;
        const int col = n0 + wn * 32 + j * 16 + (lane & 15);
        float v = acc[i][j][r] + biasj[j];
        const size_t idx = (size_t)row * N + col;
        if constexpr (EM == 0) {
          ((float*)outp)[idx] = v + resid[idx];
        } else if constexpr (EM == 1) {
          ((__bf16*)outp)[idx] = (__bf16)v;
        } else {
          ((__bf16*)outp)[idx] = (__bf16)gelu_tanh(v);
        }
      }
}

// ---- Flash attention with entropy: 1 wave per (b,h,16-query tile) ----------
// 1 wave per block with 4096 blocks: occupancy comes from block count, so give
// the wave the full VGPR budget (avoids Q-fragment scratch spills).
__global__ void __launch_bounds__(32, 1)
attn_kernel(const __bf16* __restrict__ qkv, __bf16* __restrict__ y,
            float* __restrict__ ent) {
  const int lane  = threadIdx.x;
  const int hi    = lane >> 4;
  const int qbase = blockIdx.x * 16;
  const int h     = blockIdx.y;
  const int b     = blockIdx.z;
  const int ldq   = 3 * CC;

  const __bf16* Qp = qkv + (size_t)b * TT * ldq + h * HD;
  const __bf16* Kp = Qp + CC;
  const __bf16* Vp = Qp + 2 * CC;

  const v16bf qa0 = frag_rowmajor(Qp + (size_t)qbase * ldq + 0,  ldq, lane);
  const v16bf qa1 = frag_rowmajor(Qp + (size_t)qbase * ldq + 32, ldq, lane);

  v8f o[4] = {};
  float mrow[8], lrow[8], ses[8];
#pragma unroll
  for (int r = 0; r < 8; ++r) { mrow[r] = -INFINITY; lrow[r] = 0.f; ses[r] = 0.f; }

  __shared__ __bf16 Pls[16 * 36];

  const int nkt = (qbase + 47) >> 5;
  for (int kt = 0; kt < nkt; ++kt) {
    const int k0 = kt * 32;
    float st[2][8];
#pragma unroll
    for (int t = 0; t < 2; ++t) {
      const v16bf kb0 = frag_rowmajor(Kp + (size_t)(k0 + t * 16) * ldq + 0,  ldq, lane);
      const v16bf kb1 = frag_rowmajor(Kp + (size_t)(k0 + t * 16) * ldq + 32, ldq, lane);
      v8f s = {};
      s = wmma_bf16(qa0, kb0, s);
      s = wmma_bf16(qa1, kb1, s);
      const int col = k0 + t * 16 + (lane & 15);
#pragma unroll
      for (int r = 0; r < 8; ++r) {
        const int row = qbase + r + 8 * hi;
        st[t][r] = (col <= row) ? s[r] * 0.125f : -1.0e30f;
      }
    }
    // fused online-softmax update: one pass per row keeps temporaries scalar
#pragma unroll
    for (int r = 0; r < 8; ++r) {
      float v = fmaxf(st[0][r], st[1][r]);
#pragma unroll
      for (int off = 1; off <= 8; off <<= 1) v = fmaxf(v, __shfl_xor(v, off, 32));
      const float mn = fmaxf(mrow[r], v);
      const float a  = __expf(mrow[r] - mn);
      const float x0 = __expf(st[0][r] - mn);
      const float x1 = __expf(st[1][r] - mn);
      float ps = x0 + x1;
      float pe = x0 * st[0][r] + x1 * st[1][r];
#pragma unroll
      for (int off = 1; off <= 8; off <<= 1) {
        ps += __shfl_xor(ps, off, 32);
        pe += __shfl_xor(pe, off, 32);
      }
      lrow[r] = lrow[r] * a + ps;
      ses[r]  = ses[r]  * a + pe;
      mrow[r] = mn;
      o[0][r] *= a; o[1][r] *= a; o[2][r] *= a; o[3][r] *= a;
      const int prow = r + 8 * hi;
      Pls[prow * 36 +      (lane & 15)] = (__bf16)x0;
      Pls[prow * 36 + 16 + (lane & 15)] = (__bf16)x1;
    }
    __syncthreads();
    const v16bf pa = frag_rowmajor(Pls, 36, lane);
#pragma unroll
    for (int n = 0; n < 4; ++n) {
      const v16bf vb = frag_b_strided(Vp + (size_t)k0 * ldq + n * 16, ldq, lane);
      o[n] = wmma_bf16(pa, vb, o[n]);
    }
    __syncthreads();
  }

  float inv[8];
#pragma unroll
  for (int r = 0; r < 8; ++r) inv[r] = 1.0f / lrow[r];
#pragma unroll
  for (int n = 0; n < 4; ++n)
#pragma unroll
    for (int r = 0; r < 8; ++r) {
      const int row = qbase + r + 8 * hi;
      const int col = h * HD + n * 16 + (lane & 15);
      y[((size_t)b * TT + row) * CC + col] = (__bf16)(o[n][r] * inv[r]);
    }
  if ((lane & 15) == 0) {
    float acc = 0.f;
#pragma unroll
    for (int r = 0; r < 8; ++r) acc += mrow[r] + __logf(lrow[r]) - ses[r] * inv[r];
    atomicAdd(ent, acc);
  }
}

__global__ void zero_kernel(float* p) { *p = 0.f; }

__global__ void ent_final_kernel(const float* acc, float* out) {
  *out = *acc / (float)(BB * HH * TT);
}

extern "C" void kernel_launch(void* const* d_in, const int* in_sizes, int n_in,
                              void* d_out, int out_size, void* d_ws, size_t ws_size,
                              hipStream_t stream) {
  (void)in_sizes; (void)n_in; (void)out_size; (void)ws_size;
  const float* x          = (const float*)d_in[0];
  const float* ln1_w      = (const float*)d_in[1];
  const float* ln1_b      = (const float*)d_in[2];
  const float* attn_w     = (const float*)d_in[3];
  const float* attn_b     = (const float*)d_in[4];
  const float* attnproj_w = (const float*)d_in[5];
  const float* attnproj_b = (const float*)d_in[6];
  const float* ln2_w      = (const float*)d_in[7];
  const float* ln2_b      = (const float*)d_in[8];
  const float* fc_w       = (const float*)d_in[9];
  const float* fc_b       = (const float*)d_in[10];
  const float* proj_w     = (const float*)d_in[11];
  const float* proj_b     = (const float*)d_in[12];
  float* out = (float*)d_out;

  char* ws = (char*)d_ws;
  float*  x1      = (float*)(ws);                         // 16 MB
  __bf16* qkv_bf  = (__bf16*)(ws + ((size_t)16 << 20));   // 24 MB
  __bf16* h_bf    = (__bf16*)(ws + ((size_t)40 << 20));   //  8 MB
  __bf16* y_bf    = (__bf16*)(ws + ((size_t)48 << 20));   //  8 MB
  __bf16* m_bf    = (__bf16*)(ws + ((size_t)56 << 20));   // 32 MB
  __bf16* wqkv_bf = (__bf16*)(ws + ((size_t)88 << 20));   //  6 MB
  __bf16* wap_bf  = (__bf16*)(ws + ((size_t)94 << 20));   //  2 MB
  __bf16* wfc_bf  = (__bf16*)(ws + ((size_t)96 << 20));   //  8 MB
  __bf16* wpj_bf  = (__bf16*)(ws + ((size_t)104 << 20));  //  8 MB
  float*  accum   = (float*)(ws + ((size_t)112 << 20));

  const int M = BB * TT;  // 4096

  zero_kernel<<<1, 1, 0, stream>>>(accum);
  // weights -> bf16
  wconv_kernel<<<(3 * CC * CC) / 256, 256, 0, stream>>>(attn_w, wqkv_bf, 3 * CC * CC);
  wconv_kernel<<<(CC * CC) / 256, 256, 0, stream>>>(attnproj_w, wap_bf, CC * CC);
  wconv_kernel<<<(4 * CC * CC) / 256, 256, 0, stream>>>(fc_w, wfc_bf, 4 * CC * CC);
  wconv_kernel<<<(4 * CC * CC) / 256, 256, 0, stream>>>(proj_w, wpj_bf, 4 * CC * CC);
  // h = LN1(x)
  ln_kernel<<<M, 256, 0, stream>>>(x, ln1_w, ln1_b, h_bf);
  // qkv = h @ attn_w^T + attn_b -> bf16 [4096, 3072]
  gemm_kernel<1><<<dim3(3 * CC / 128, M / 128), 256, 0, stream>>>(
      h_bf, wqkv_bf, attn_b, nullptr, qkv_bf, M, 3 * CC, CC);
  // y = softmax(QK^T/sqrt(HD)) V (+ entropy), bf16 out
  attn_kernel<<<dim3(TT / 16, HH, BB), 32, 0, stream>>>(qkv_bf, y_bf, accum);
  // x1 = x + y @ attnproj_w^T + b (f32)
  gemm_kernel<0><<<dim3(CC / 128, M / 128), 256, 0, stream>>>(
      y_bf, wap_bf, attnproj_b, x, x1, M, CC, CC);
  // h2 = LN2(x1)
  ln_kernel<<<M, 256, 0, stream>>>(x1, ln2_w, ln2_b, h_bf);
  // m = gelu(h2 @ fc_w^T + fc_b) -> bf16 [4096, 4096], fused epilogue
  gemm_kernel<2><<<dim3(4 * CC / 128, M / 128), 256, 0, stream>>>(
      h_bf, wfc_bf, fc_b, nullptr, m_bf, M, 4 * CC, CC);
  // out = x1 + m @ proj_w^T + proj_b (f32)
  gemm_kernel<0><<<dim3(CC / 128, M / 128), 256, 0, stream>>>(
      m_bf, wpj_bf, proj_b, x1, out, M, CC, 4 * CC);
  // entropy scalar
  ent_final_kernel<<<1, 1, 0, stream>>>(accum, out + (size_t)M * CC);
}
